// TriangleMul_26293789786411
// MI455X (gfx1250) — matrix operand order repaired and checked by reading, not compiled, not run on hardware
//
#include <hip/hip_runtime.h>
#include <cstdint>
#include <cstddef>

typedef __bf16 bf16;
typedef __attribute__((ext_vector_type(16))) bf16  v16bf;
typedef __attribute__((ext_vector_type(8)))  bf16  v8bf;
typedef __attribute__((ext_vector_type(8)))  float v8f;

#define NSEQ 512
#define DCH  128
#define NPOS (NSEQ * NSEQ)   // 262144 edge positions

// ---------------------------------------------------------------------------
// helpers
// ---------------------------------------------------------------------------
__device__ __forceinline__ float wave_sum(float v) {
#pragma unroll
  for (int off = 16; off > 0; off >>= 1) v += __shfl_xor(v, off, 32);
  return v;
}

__device__ __forceinline__ float sigmoidf(float x) {
  return 1.0f / (1.0f + __expf(-x));
}

// A fragment: 16x32 bf16 tile, source rows are k-contiguous with leading dim ld.
// Per ISA 7.12.2: lane m=L&15 holds row M=m; lanes<16 hold K {0..7,16..23},
// lanes>=16 hold K {8..15,24..31} (within the 32-wide K window at kb).
__device__ __forceinline__ v16bf load_a_frag(const bf16* base, int ld, int row0,
                                             int kb, int lane) {
  const int m  = lane & 15;
  const int hi = lane >> 4;
  const bf16* rp = base + (row0 + m) * ld;
  v8bf lo = *(const v8bf*)(rp + kb + hi * 8);
  v8bf hh = *(const v8bf*)(rp + kb + 16 + hi * 8);
  return __builtin_shufflevector(lo, hh, 0, 1, 2, 3, 4, 5, 6, 7,
                                         8, 9, 10, 11, 12, 13, 14, 15);
}

// B fragment: 32x16 bf16 tile of B^T where source rows (n) are k-contiguous.
// Lane n=L&15 holds column N=n; lanes<16 hold K {0..7,16..23}, lanes>=16 hold K {8..15,24..31}.
__device__ __forceinline__ v16bf load_b_frag(const bf16* base, int ld, int col0,
                                             int kb, int lane) {
  const int n  = lane & 15;
  const int kh = (lane >> 4) * 8;
  const bf16* rp = base + (col0 + n) * ld + kb + kh;
  v8bf lo = *(const v8bf*)(rp);
  v8bf hh = *(const v8bf*)(rp + 16);
  return __builtin_shufflevector(lo, hh, 0, 1, 2, 3, 4, 5, 6, 7,
                                         8, 9, 10, 11, 12, 13, 14, 15);
}

__device__ __forceinline__ v8f wmma_bf16f32(v16bf a, v16bf b, v8f c) {
  return __builtin_amdgcn_wmma_f32_16x16x32_bf16(false, a, false, b,
                                                 (short)0, c, false, false);
}

// CDNA5 async global->LDS copy, 16B per lane, tracked by ASYNCcnt.
// lds_byte_off: wave-relative LDS byte address (low 32 bits of generic ptr).
__device__ __forceinline__ void async_load_16B(uint32_t lds_byte_off,
                                               const void* gaddr) {
  asm volatile("global_load_async_to_lds_b128 %0, %1, off"
               :
               : "v"(lds_byte_off), "v"(gaddr)
               : "memory");
}

template <int N>
__device__ __forceinline__ void wait_asynccnt() {
  asm volatile("s_wait_asynccnt %0" : : "i"(N) : "memory");
}

__device__ __forceinline__ uint32_t lds_off(const void* p) {
  return (uint32_t)(uintptr_t)p;  // LDS aperture: addr[31:0] == LDS byte offset
}

// ---------------------------------------------------------------------------
// kernel 0: weight precast f32 -> bf16.  Wcat = [W_fg(512) ; W_go(128)] x 128
// ---------------------------------------------------------------------------
__global__ __launch_bounds__(256) void k_prep(const float* __restrict__ W_fg,
                                              const float* __restrict__ W_go,
                                              const float* __restrict__ W_out,
                                              bf16* __restrict__ Wcat,
                                              bf16* __restrict__ WoutB) {
  const int i = blockIdx.x * 256 + threadIdx.x;
  if (i < 512 * DCH) {
    Wcat[i] = (bf16)W_fg[i];
  } else if (i < 640 * DCH) {
    Wcat[i] = (bf16)W_go[i - 512 * DCH];
  } else if (i < 640 * DCH + DCH * DCH) {
    const int j = i - 640 * DCH;
    WoutB[j] = (bf16)W_out[j];
  }
}

// ---------------------------------------------------------------------------
// kernel 1: LN1 + fused projections + gating + mask.
// Per block: 64 positions.  Writes To_T[d][p], From_T[d][p] (bf16, transposed)
// and out_gate[p][d] (f32).
// ---------------------------------------------------------------------------
__global__ __launch_bounds__(256) void k_ln_proj(
    const float* __restrict__ edges, const unsigned char* __restrict__ mask,
    const float* __restrict__ ln1w, const float* __restrict__ ln1b,
    const bf16* __restrict__ Wcat, const float* __restrict__ b_fg,
    const float* __restrict__ b_go, bf16* __restrict__ ToT,
    bf16* __restrict__ FromT, float* __restrict__ og) {
  __shared__ __align__(16) char pool[17408 * 3];
  bf16(*As)[136]  = (bf16(*)[136])(pool);              // 64 x 128 bf16 (pad 136)
  bf16(*Bsf)[136] = (bf16(*)[136])(pool + 17408);      // feat/go weight chunk
  bf16(*Bsg)[136] = (bf16(*)[136])(pool + 2 * 17408);  // gate weight chunk
  bf16(*Ts)[72]   = (bf16(*)[72])(pool + 17408);       // epilogue overlay
  float(*Gs)[72]  = (float(*)[72])(pool + 17408);      // epilogue overlay

  const int t = threadIdx.x, lane = t & 31, wv = t >> 5;
  const int p0 = blockIdx.x * 64;

  // ---- LayerNorm(edges[p0..p0+63, :]) -> As bf16 ----
  {
    const float4 w4 = *(const float4*)(ln1w + lane * 4);
    const float4 b4 = *(const float4*)(ln1b + lane * 4);
#pragma unroll
    for (int q = 0; q < 8; ++q) {
      const int m = wv * 8 + q;
      const float4 x4 =
          *(const float4*)(edges + (size_t)(p0 + m) * DCH + lane * 4);
      float s  = x4.x + x4.y + x4.z + x4.w;
      float s2 = x4.x * x4.x + x4.y * x4.y + x4.z * x4.z + x4.w * x4.w;
      s  = wave_sum(s);
      s2 = wave_sum(s2);
      const float mu = s * (1.0f / 128.0f);
      const float rs = rsqrtf(s2 * (1.0f / 128.0f) - mu * mu + 1e-5f);
      As[m][lane * 4 + 0] = (bf16)((x4.x - mu) * rs * w4.x + b4.x);
      As[m][lane * 4 + 1] = (bf16)((x4.y - mu) * rs * w4.y + b4.y);
      As[m][lane * 4 + 2] = (bf16)((x4.z - mu) * rs * w4.z + b4.z);
      As[m][lane * 4 + 3] = (bf16)((x4.w - mu) * rs * w4.w + b4.w);
    }
  }

  const int r  = wv >> 1;        // C row subtile 0..3
  const int c0 = (wv & 1) * 2;   // C col subtiles {0,1} or {2,3}

  for (int c = 0; c < 6; ++c) {
    const bool paired = (c < 4);                         // feat+gate pair
    const int rowF = paired ? (c * 64) : (512 + (c - 4) * 64);
    __syncthreads();  // prior staging done before Bs overwrite
#pragma unroll
    for (int it = 0; it < 4; ++it) {
      const int idx = t + it * 256;
      const int row = idx >> 4, seg = idx & 15;
      *(uint4*)(&Bsf[row][seg * 8]) =
          *(const uint4*)(Wcat + (size_t)(rowF + row) * DCH + seg * 8);
    }
    if (paired) {
      const int rowG = 256 + c * 64;
#pragma unroll
      for (int it = 0; it < 4; ++it) {
        const int idx = t + it * 256;
        const int row = idx >> 4, seg = idx & 15;
        *(uint4*)(&Bsg[row][seg * 8]) =
            *(const uint4*)(Wcat + (size_t)(rowG + row) * DCH + seg * 8);
      }
    }
    __syncthreads();

    v8f f0 = {}, f1 = {}, g0 = {}, g1 = {};
#pragma unroll
    for (int kb = 0; kb < 128; kb += 32) {
      const v16bf a   = load_a_frag(&As[0][0], 136, r * 16, kb, lane);
      const v16bf bf0 = load_b_frag(&Bsf[0][0], 136, (c0 + 0) * 16, kb, lane);
      const v16bf bf1 = load_b_frag(&Bsf[0][0], 136, (c0 + 1) * 16, kb, lane);
      f0 = wmma_bf16f32(a, bf0, f0);
      f1 = wmma_bf16f32(a, bf1, f1);
      if (paired) {
        const v16bf bg0 = load_b_frag(&Bsg[0][0], 136, (c0 + 0) * 16, kb, lane);
        const v16bf bg1 = load_b_frag(&Bsg[0][0], 136, (c0 + 1) * 16, kb, lane);
        g0 = wmma_bf16f32(a, bg0, g0);
        g1 = wmma_bf16f32(a, bg1, g1);
      }
    }
    __syncthreads();  // Bs free -> staging overlay legal

    const int n0 = c0 * 16 + (lane & 15);
    const int n1 = n0 + 16;
    const int mbase = r * 16 + (lane >> 4) * 8;
    if (paired) {
      const float bff0 = b_fg[rowF + n0], bff1 = b_fg[rowF + n1];
      const float bfg0 = b_fg[256 + c * 64 + n0];
      const float bfg1 = b_fg[256 + c * 64 + n1];
#pragma unroll
      for (int rr = 0; rr < 8; ++rr) {
        const int m = mbase + rr;
        const float mk = mask[p0 + m] ? 1.0f : 0.0f;
        Ts[n0][m] = (bf16)((f0[rr] + bff0) * sigmoidf(g0[rr] + bfg0) * mk);
        Ts[n1][m] = (bf16)((f1[rr] + bff1) * sigmoidf(g1[rr] + bfg1) * mk);
      }
      __syncthreads();
      bf16* dst = ((c < 2) ? ToT : FromT) + (size_t)((c & 1) * 64) * NPOS + p0;
#pragma unroll
      for (int it = 0; it < 2; ++it) {
        const int idx = t + it * 256;
        const int row = idx >> 3, seg = idx & 7;
        *(uint4*)(dst + (size_t)row * NPOS + seg * 8) =
            *(const uint4*)(&Ts[row][seg * 8]);
      }
    } else {
      const int ddb = (c - 4) * 64;
      const float bg0 = b_go[ddb + n0], bg1 = b_go[ddb + n1];
#pragma unroll
      for (int rr = 0; rr < 8; ++rr) {
        const int m = mbase + rr;
        const float mk = mask[p0 + m] ? 1.0f : 0.0f;
        Gs[m][n0] = sigmoidf(f0[rr] + bg0) * mk;
        Gs[m][n1] = sigmoidf(f1[rr] + bg1) * mk;
      }
      __syncthreads();
#pragma unroll
      for (int it = 0; it < 4; ++it) {
        const int idx = t + it * 256;
        const int row = idx >> 4, seg = idx & 15;
        *(float4*)(og + (size_t)(p0 + row) * DCH + ddb + seg * 4) =
            *(const float4*)(&Gs[row][seg * 4]);
      }
    }
  }
}

// ---------------------------------------------------------------------------
// kernel 2: tri[i,j,d] = sum_k to[i,k,d]*from[j,k,d]  (per-channel GEMM)
// Block: one 64x64 C tile of one channel, K=512.
// Double-buffered via CDNA5 async global->LDS loads (ASYNCcnt).
// ---------------------------------------------------------------------------
#define KTRI_BUF 18432  // As(9216) + Bs(9216) per buffer

__device__ __forceinline__ void k_tri_issue_chunk(const bf16* Ab, const bf16* Bb,
                                                  char* ldsbase, int kk, int t) {
  // 64 rows x 64 bf16 per matrix = 512 x 16B chunks; 256 threads -> 2 each
#pragma unroll
  for (int it = 0; it < 2; ++it) {
    const int idx = t + it * 256;
    const int row = idx >> 3, seg = idx & 7;
    async_load_16B(lds_off(ldsbase + row * 144 + seg * 16),
                   Ab + (size_t)row * NSEQ + kk + seg * 8);
    async_load_16B(lds_off(ldsbase + 9216 + row * 144 + seg * 16),
                   Bb + (size_t)row * NSEQ + kk + seg * 8);
  }
}

__global__ __launch_bounds__(256) void k_tri(const bf16* __restrict__ ToT,
                                             const bf16* __restrict__ FromT,
                                             float* __restrict__ tri) {
  __shared__ __align__(16) char pool[2 * KTRI_BUF];  // ping-pong As+Bs
  float(*Cs)[68] = (float(*)[68])(pool);             // epilogue overlay 64x68 f32

  const int t = threadIdx.x, lane = t & 31, wv = t >> 5;
  const int d  = blockIdx.z;
  const int i0 = blockIdx.y * 64;
  const int j0 = blockIdx.x * 64;
  const bf16* Ab = ToT + (size_t)d * NPOS + (size_t)i0 * NSEQ;
  const bf16* Bb = FromT + (size_t)d * NPOS + (size_t)j0 * NSEQ;

  const int r  = wv >> 1;
  const int c0 = (wv & 1) * 2;
  v8f acc0 = {}, acc1 = {};

  // prologue: fill both buffers (4 async ops / thread / chunk)
  k_tri_issue_chunk(Ab, Bb, pool, 0, t);
  k_tri_issue_chunk(Ab, Bb, pool + KTRI_BUF, 64, t);

#pragma unroll
  for (int i = 0; i < 8; ++i) {
    char* cur = pool + (i & 1) * KTRI_BUF;
    // chunk i done when <=4 of this thread's async ops remain (in-order);
    // last chunk has nothing behind it -> drain fully.
    if (i < 7) wait_asynccnt<4>(); else wait_asynccnt<0>();
    __syncthreads();  // all threads' chunk-i data resident in LDS
    const bf16* Asb = (const bf16*)cur;
    const bf16* Bsb = (const bf16*)(cur + 9216);
#pragma unroll
    for (int kb = 0; kb < 64; kb += 32) {
      const v16bf a  = load_a_frag(Asb, 72, r * 16, kb, lane);
      const v16bf b0 = load_b_frag(Bsb, 72, (c0 + 0) * 16, kb, lane);
      const v16bf b1 = load_b_frag(Bsb, 72, (c0 + 1) * 16, kb, lane);
      acc0 = wmma_bf16f32(a, b0, acc0);
      acc1 = wmma_bf16f32(a, b1, acc1);
    }
    __syncthreads();  // everyone done reading `cur` before refilling it
    if (i + 2 < 8) k_tri_issue_chunk(Ab, Bb, cur, (i + 2) * 64, t);
  }

  const int n0 = c0 * 16 + (lane & 15);
  const int mbase = r * 16 + (lane >> 4) * 8;
#pragma unroll
  for (int rr = 0; rr < 8; ++rr) {
    Cs[mbase + rr][n0]      = acc0[rr];
    Cs[mbase + rr][n0 + 16] = acc1[rr];
  }
  __syncthreads();
  float* dst = tri + (size_t)d * NPOS + (size_t)i0 * NSEQ + j0;
#pragma unroll
  for (int it = 0; it < 4; ++it) {
    const int idx = t + it * 256;
    const int row = idx >> 4, seg = idx & 15;
    *(float4*)(dst + (size_t)row * NSEQ + seg * 4) =
        *(const float4*)(&Cs[row][seg * 4]);
  }
}

// ---------------------------------------------------------------------------
// kernel 3: out = out_gate * (LN2(tri) @ W_out^T + b_out)
// Block: 64 positions.  Gathers tri[d][p] -> LDS [p][d], LN2, 64x128x128 GEMM.
// ---------------------------------------------------------------------------
__global__ __launch_bounds__(256) void k_out(
    const float* __restrict__ tri, const float* __restrict__ og,
    const float* __restrict__ ln2w, const float* __restrict__ ln2b,
    const bf16* __restrict__ WoutB, const float* __restrict__ b_out,
    float* __restrict__ out) {
  __shared__ __align__(16) float trile[64][132];  // also C staging
  __shared__ __align__(16) bf16 xt[64][136];

  const int t = threadIdx.x, lane = t & 31, wv = t >> 5;
  const int p0 = blockIdx.x * 64;

  // gather + transpose tri tile into LDS [pos][d]
#pragma unroll
  for (int it = 0; it < 8; ++it) {
    const int idx = t + it * 256;
    const int d = idx >> 4, seg = idx & 15;
    const float4 v = *(const float4*)(tri + (size_t)d * NPOS + p0 + seg * 4);
    trile[seg * 4 + 0][d] = v.x;
    trile[seg * 4 + 1][d] = v.y;
    trile[seg * 4 + 2][d] = v.z;
    trile[seg * 4 + 3][d] = v.w;
  }
  __syncthreads();

  // LN2 -> xt bf16
  {
    const float4 w4 = *(const float4*)(ln2w + lane * 4);
    const float4 b4 = *(const float4*)(ln2b + lane * 4);
#pragma unroll
    for (int q = 0; q < 8; ++q) {
      const int m = wv * 8 + q;
      const float x0 = trile[m][lane * 4 + 0];
      const float x1 = trile[m][lane * 4 + 1];
      const float x2 = trile[m][lane * 4 + 2];
      const float x3 = trile[m][lane * 4 + 3];
      float s  = x0 + x1 + x2 + x3;
      float s2 = x0 * x0 + x1 * x1 + x2 * x2 + x3 * x3;
      s  = wave_sum(s);
      s2 = wave_sum(s2);
      const float mu = s * (1.0f / 128.0f);
      const float rs = rsqrtf(s2 * (1.0f / 128.0f) - mu * mu + 1e-5f);
      xt[m][lane * 4 + 0] = (bf16)((x0 - mu) * rs * w4.x + b4.x);
      xt[m][lane * 4 + 1] = (bf16)((x1 - mu) * rs * w4.y + b4.y);
      xt[m][lane * 4 + 2] = (bf16)((x2 - mu) * rs * w4.z + b4.z);
      xt[m][lane * 4 + 3] = (bf16)((x3 - mu) * rs * w4.w + b4.w);
    }
  }
  __syncthreads();

  // GEMM: 64 x 128 x 128, B frags straight from L2-resident bf16 W_out
  const int r  = wv & 3;
  const int cg = wv >> 2;
  v8f acc[4] = {{}, {}, {}, {}};
#pragma unroll
  for (int kb = 0; kb < 128; kb += 32) {
    const v16bf a = load_a_frag(&xt[0][0], 136, r * 16, kb, lane);
#pragma unroll
    for (int u = 0; u < 4; ++u) {
      const v16bf b = load_b_frag(WoutB, DCH, (cg * 4 + u) * 16, kb, lane);
      acc[u] = wmma_bf16f32(a, b, acc[u]);
    }
  }
  __syncthreads();  // trile free -> reuse as C staging
  float(*Cs)[132] = trile;
  const int nn = lane & 15;
  const int mbase = r * 16 + (lane >> 4) * 8;
#pragma unroll
  for (int u = 0; u < 4; ++u) {
    const int col = cg * 64 + u * 16 + nn;
#pragma unroll
    for (int rr = 0; rr < 8; ++rr) Cs[mbase + rr][col] = acc[u][rr];
  }
  __syncthreads();
#pragma unroll
  for (int it = 0; it < 8; ++it) {
    const int idx = t + it * 256;
    const int row = idx >> 5, seg = idx & 31;
    const float4 cv = *(const float4*)(&Cs[row][seg * 4]);
    const float4 bv = *(const float4*)(b_out + seg * 4);
    const float4 gv = *(const float4*)(og + (size_t)(p0 + row) * DCH + seg * 4);
    float4 o;
    o.x = gv.x * (cv.x + bv.x);
    o.y = gv.y * (cv.y + bv.y);
    o.z = gv.z * (cv.z + bv.z);
    o.w = gv.w * (cv.w + bv.w);
    *(float4*)(out + (size_t)(p0 + row) * DCH + seg * 4) = o;
  }
}

// ---------------------------------------------------------------------------
// launcher
// ---------------------------------------------------------------------------
extern "C" void kernel_launch(void* const* d_in, const int* in_sizes, int n_in,
                              void* d_out, int out_size, void* d_ws,
                              size_t ws_size, hipStream_t stream) {
  const float* edges        = (const float*)d_in[0];
  const unsigned char* mask = (const unsigned char*)d_in[1];
  const float* ln1w  = (const float*)d_in[2];
  const float* ln1b  = (const float*)d_in[3];
  const float* W_fg  = (const float*)d_in[4];
  const float* b_fg  = (const float*)d_in[5];
  const float* W_go  = (const float*)d_in[6];
  const float* b_go  = (const float*)d_in[7];
  const float* ln2w  = (const float*)d_in[8];
  const float* ln2b  = (const float*)d_in[9];
  const float* W_out = (const float*)d_in[10];
  const float* b_out = (const float*)d_in[11];
  float* out = (float*)d_out;

  char* ws = (char*)d_ws;
  // workspace layout (bytes):
  //   To_T   bf16 [128][262144]   67,108,864
  //   From_T bf16 [128][262144]   67,108,864
  //   og     f32  [262144][128]  134,217,728
  //   tri    f32  [128][262144]  134,217,728
  //   Wcat   bf16 [640][128]         163,840
  //   WoutB  bf16 [128][128]          32,768
  bf16*  ToT   = (bf16*)(ws);
  bf16*  FromT = (bf16*)(ws + 67108864);
  float* og    = (float*)(ws + 134217728);
  float* tri   = (float*)(ws + 268435456);
  bf16*  Wcat  = (bf16*)(ws + 402653184);
  bf16*  WoutB = (bf16*)(ws + 402817024);

  k_prep<<<384, 256, 0, stream>>>(W_fg, W_go, W_out, Wcat, WoutB);
  k_ln_proj<<<NPOS / 64, 256, 0, stream>>>(edges, mask, ln1w, ln1b, Wcat, b_fg,
                                           b_go, ToT, FromT, og);
  k_tri<<<dim3(NSEQ / 64, NSEQ / 64, DCH), 256, 0, stream>>>(ToT, FromT, tri);
  k_out<<<NPOS / 64, 256, 0, stream>>>(tri, og, ln2w, ln2b, WoutB, b_out, out);
}
